// CausalSelfAttention_55138790146376
// MI455X (gfx1250) — matrix operand order repaired and checked by reading, not compiled, not run on hardware
//
#include <hip/hip_runtime.h>

#define D_MODEL 1024
#define NHEAD   16
#define SEQ     2048
#define BATCH   4
#define HDIM    64   // D_MODEL / NHEAD

typedef __attribute__((ext_vector_type(16))) __bf16 v16bf;
typedef __attribute__((ext_vector_type(8)))  __bf16 v8bf;
typedef __attribute__((ext_vector_type(4)))  __bf16 v4bf;
typedef __attribute__((ext_vector_type(8)))  float  v8f;

// ---------------------------------------------------------------------------
// fp32 -> bf16 conversion (vectorized 4-wide)
// ---------------------------------------------------------------------------
__global__ void cvt_bf16_kernel(const float* __restrict__ s, __bf16* __restrict__ d, int n4) {
  int i = blockIdx.x * blockDim.x + threadIdx.x;
  if (i < n4) {
    const float4 v = ((const float4*)s)[i];
    v4bf o;
    o[0] = (__bf16)v.x; o[1] = (__bf16)v.y; o[2] = (__bf16)v.z; o[3] = (__bf16)v.w;
    ((v4bf*)d)[i] = o;
  }
}

// fp32 [K][N] -> bf16 transposed [N][K]; one thread per 8-run of k.
__global__ void cvt_transpose_bf16_kernel(const float* __restrict__ w,
                                          __bf16* __restrict__ wt, int K, int N) {
  const int K8 = K >> 3;
  int idx = blockIdx.x * blockDim.x + threadIdx.x;
  if (idx >= N * K8) return;
  const int n = idx / K8, k0 = (idx - n * K8) << 3;
  v8bf o;
#pragma unroll
  for (int j = 0; j < 8; ++j) o[j] = (__bf16)w[(size_t)(k0 + j) * N + n];
  *(v8bf*)&wt[(size_t)n * K + k0] = o;
}

// qkv [B*T][3072] -> VT [B][H][hd][T] (keys contiguous); thread = 8-run of t.
__global__ void repack_vt_kernel(const __bf16* __restrict__ qkv, __bf16* __restrict__ vt) {
  int idx = blockIdx.x * blockDim.x + threadIdx.x;           // B*H*HDIM*(SEQ/8) = 1,048,576
  const int t0 = (idx & (SEQ / 8 - 1)) << 3;
  const int d  = (idx >> 8) & (HDIM - 1);
  const int h  = (idx >> 14) & (NHEAD - 1);
  const int b  = idx >> 18;
  v8bf o;
#pragma unroll
  for (int j = 0; j < 8; ++j)
    o[j] = qkv[(size_t)(b * SEQ + t0 + j) * (3 * D_MODEL) + 2 * D_MODEL + h * HDIM + d];
  *(v8bf*)&vt[((size_t)(b * NHEAD + h) * HDIM + d) * SEQ + t0] = o;
}

// ---------------------------------------------------------------------------
// WMMA fragment loaders (layouts per CDNA5 ISA 7.12.2, wave32).
// All element runs are contiguous so the compiler fuses them into b128 loads.
// ---------------------------------------------------------------------------
// A-matrix 16x32 bf16, row-major source, row stride ld (elements).
__device__ __forceinline__ v16bf load_a_frag(const __bf16* base, int ld) {
  const int lane = threadIdx.x & 31;
  const int m  = lane & 15;
  const int kh = lane >> 4;
  v16bf a;
#pragma unroll
  for (int e = 0; e < 16; ++e) {
    const int g = e >> 3, j = e & 7;
    a[e] = base[(size_t)m * ld + g * 16 + kh * 8 + j];    // two contiguous 8-runs
  }
  return a;
}

// B-matrix 32x16 where SOURCE is row-major [N][K] (transposed operand):
// two contiguous 8-element runs per lane (k = g*16 + kh*8 + j) -> 2x b128.
__device__ __forceinline__ v16bf load_bT_frag(const __bf16* base, int ld) {
  const int lane = threadIdx.x & 31;
  const int n  = lane & 15;
  const int kh = lane >> 4;
  v16bf b;
#pragma unroll
  for (int e = 0; e < 16; ++e) b[e] = base[(size_t)n * ld + (e >> 3) * 16 + kh * 8 + (e & 7)];
  return b;
}

// ---------------------------------------------------------------------------
// GEMM: C[M,N] = A[M,K](bf16 row-major) @ BT[N,K](bf16 row-major, = B^T)
// No LDS. block = 128 threads = 4 waves (2x2), wave tile 64x64:
// per BK=32 step -> 16 WMMAs : 16 b128 loads (1:1).
// ---------------------------------------------------------------------------
template <bool OUT_F32>
__global__ __launch_bounds__(128) void gemm_bf16_wmma(
    const __bf16* __restrict__ A, const __bf16* __restrict__ BT,
    void* __restrict__ C, int M, int N, int K) {
  const int wave = threadIdx.x >> 5;
  const int lane = threadIdx.x & 31;
  const int wm = wave >> 1, wn = wave & 1;
  const int row0 = blockIdx.y * 128 + wm * 64;
  const int col0 = blockIdx.x * 128 + wn * 64;

  v8f acc[4][4];
#pragma unroll
  for (int mi = 0; mi < 4; ++mi)
#pragma unroll
    for (int ni = 0; ni < 4; ++ni)
#pragma unroll
      for (int r = 0; r < 8; ++r) acc[mi][ni][r] = 0.0f;

  for (int k0 = 0; k0 < K; k0 += 32) {
    if (k0 + 32 < K)
      __builtin_prefetch(A + (size_t)row0 * K + k0 + 32, 0, 1);  // global_prefetch_b8

    v16bf af[4];
#pragma unroll
    for (int mi = 0; mi < 4; ++mi)
      af[mi] = load_a_frag(A + (size_t)(row0 + mi * 16) * K + k0, K);
#pragma unroll
    for (int ni = 0; ni < 4; ++ni) {
      const v16bf bfr = load_bT_frag(BT + (size_t)(col0 + ni * 16) * K + k0, K);
#pragma unroll
      for (int mi = 0; mi < 4; ++mi)
        acc[mi][ni] = __builtin_amdgcn_wmma_f32_16x16x32_bf16(
            false, af[mi], false, bfr, (short)0, acc[mi][ni], false, false);
    }
  }

  // C/D layout: row M = (lane>>4)*8 + r, col N = lane & 15
  const int cn = lane & 15, mh = lane >> 4;
#pragma unroll
  for (int mi = 0; mi < 4; ++mi)
#pragma unroll
    for (int ni = 0; ni < 4; ++ni)
#pragma unroll
      for (int r = 0; r < 8; ++r) {
        const size_t gr = (size_t)(row0 + mi * 16 + mh * 8 + r);
        const size_t gc = (size_t)(col0 + ni * 16 + cn);
        if (OUT_F32) ((float*)C)[gr * N + gc] = acc[mi][ni][r];
        else         ((__bf16*)C)[gr * N + gc] = (__bf16)acc[mi][ni][r];
      }
}

// ---------------------------------------------------------------------------
// Flash attention: 1 wave per *32* queries (two 16-row tiles), online softmax,
// 32-key blocks. K/V fragments are loaded once per block and feed both query
// tiles: 16 b128 loads -> 18 WMMAs per block iteration.
// qkv: [B*T][3072] (Q @ 0, K @ 1024); V pre-transposed: VT [B][H][hd][T].
// Row sums of P via matrix engine (P @ ones broadcasts row sums per lane).
// ---------------------------------------------------------------------------
__global__ __launch_bounds__(128) void flash_attn_wmma(
    const __bf16* __restrict__ qkv, const __bf16* __restrict__ vt,
    __bf16* __restrict__ y) {
  __shared__ __bf16 Pl[8][16][40];  // per (wave, qt) P staging

  const int tid  = threadIdx.x;
  const int wave = tid >> 5, lane = tid & 31;
  const int cn = lane & 15, mh = lane >> 4;
  const int blocks_per_bh = SEQ / 128;                    // 16
  const int bh = blockIdx.x / blocks_per_bh;
  const int b = bh / NHEAD, h = bh % NHEAD;
  const int q0 = (blockIdx.x % blocks_per_bh) * 128 + wave * 32;

  const size_t ROW = 3 * D_MODEL;
  const __bf16* base  = qkv + (size_t)b * SEQ * ROW;
  const __bf16* vhead = vt + (size_t)(b * NHEAD + h) * HDIM * SEQ;

  // Q fragments: 2 tiles x 2 K-halves
  v16bf qf[2][2];
#pragma unroll
  for (int qt = 0; qt < 2; ++qt)
#pragma unroll
    for (int hf = 0; hf < 2; ++hf)
      qf[qt][hf] = load_a_frag(
          base + (size_t)(q0 + qt * 16) * ROW + h * HDIM + hf * 32, (int)ROW);

  // all-ones B fragment for row-sum WMMA
  v16bf onesf;
#pragma unroll
  for (int e = 0; e < 16; ++e) onesf[e] = (__bf16)1.0f;

  float mrow[2][8], lrow[2][8], alphav[2][8];
  v8f yacc[2][4];
#pragma unroll
  for (int qt = 0; qt < 2; ++qt) {
#pragma unroll
    for (int r = 0; r < 8; ++r) { mrow[qt][r] = -3.0e38f; lrow[qt][r] = 0.0f; }
#pragma unroll
    for (int nd = 0; nd < 4; ++nd)
#pragma unroll
      for (int r = 0; r < 8; ++r) yacc[qt][nd][r] = 0.0f;
  }

  const float scale = 0.125f;               // 1/sqrt(64)
  const int nkb = q0 / 32 + 1;              // key blocks covering keys 0..q0+31

  for (int kb = 0; kb < nkb; ++kb) {
    const int key0 = kb * 32;

    // S[qt] = Q[qt] @ K^T : K fragments shared across both query tiles
    v8f S[2][2];
#pragma unroll
    for (int qt = 0; qt < 2; ++qt)
#pragma unroll
      for (int nb = 0; nb < 2; ++nb)
#pragma unroll
        for (int r = 0; r < 8; ++r) S[qt][nb][r] = 0.0f;
#pragma unroll
    for (int nb = 0; nb < 2; ++nb)
#pragma unroll
      for (int hf = 0; hf < 2; ++hf) {
        const v16bf kf = load_bT_frag(
            base + (size_t)(key0 + nb * 16) * ROW + D_MODEL + h * HDIM + hf * 32, (int)ROW);
#pragma unroll
        for (int qt = 0; qt < 2; ++qt)
          S[qt][nb] = __builtin_amdgcn_wmma_f32_16x16x32_bf16(
              false, qf[qt][hf], false, kf, (short)0, S[qt][nb], false, false);
      }

    // online softmax per tile: mask, row-max (shuffle reduce), exp, stage P
#pragma unroll
    for (int qt = 0; qt < 2; ++qt)
#pragma unroll
      for (int r = 0; r < 8; ++r) {
        const int qg = q0 + qt * 16 + mh * 8 + r;
        float s0 = S[qt][0][r] * scale;
        float s1 = S[qt][1][r] * scale;
        s0 = (key0 + cn      <= qg) ? s0 : -3.0e38f;
        s1 = (key0 + 16 + cn <= qg) ? s1 : -3.0e38f;
        float mx = fmaxf(s0, s1);
#pragma unroll
        for (int dd = 1; dd < 16; dd <<= 1) mx = fmaxf(mx, __shfl_xor(mx, dd, 32));
        const float mnew  = fmaxf(mrow[qt][r], mx);
        const float alpha = __expf(mrow[qt][r] - mnew);
        const float p0 = __expf(s0 - mnew);
        const float p1 = __expf(s1 - mnew);
        mrow[qt][r]   = mnew;
        alphav[qt][r] = alpha;
        Pl[wave * 2 + qt][mh * 8 + r][cn]      = (__bf16)p0;
        Pl[wave * 2 + qt][mh * 8 + r][16 + cn] = (__bf16)p1;
#pragma unroll
        for (int nd = 0; nd < 4; ++nd) yacc[qt][nd][r] *= alpha;
      }

    // P tiles as A-fragments (from LDS)
    v16bf pf[2];
#pragma unroll
    for (int qt = 0; qt < 2; ++qt)
      pf[qt] = load_a_frag(&Pl[wave * 2 + qt][0][0], 40);

    // row sums via matrix engine
#pragma unroll
    for (int qt = 0; qt < 2; ++qt) {
      v8f sums;
#pragma unroll
      for (int r = 0; r < 8; ++r) sums[r] = 0.0f;
      sums = __builtin_amdgcn_wmma_f32_16x16x32_bf16(false, pf[qt], false, onesf,
                                                     (short)0, sums, false, false);
#pragma unroll
      for (int r = 0; r < 8; ++r) lrow[qt][r] = lrow[qt][r] * alphav[qt][r] + sums[r];
    }

    // yacc += P @ V : V fragments shared across both query tiles
#pragma unroll
    for (int nd = 0; nd < 4; ++nd) {
      const v16bf vf = load_bT_frag(vhead + (size_t)(nd * 16) * SEQ + key0, SEQ);
#pragma unroll
      for (int qt = 0; qt < 2; ++qt)
        yacc[qt][nd] = __builtin_amdgcn_wmma_f32_16x16x32_bf16(
            false, pf[qt], false, vf, (short)0, yacc[qt][nd], false, false);
    }
  }

  // epilogue: y = yacc / l -> bf16 [B*T][D] head slice
#pragma unroll
  for (int qt = 0; qt < 2; ++qt)
#pragma unroll
    for (int nd = 0; nd < 4; ++nd)
#pragma unroll
      for (int r = 0; r < 8; ++r) {
        const float v = yacc[qt][nd][r] / lrow[qt][r];
        y[(size_t)(b * SEQ + q0 + qt * 16 + mh * 8 + r) * D_MODEL +
          h * HDIM + nd * 16 + cn] = (__bf16)v;
      }
}

// ---------------------------------------------------------------------------
// launch
// ---------------------------------------------------------------------------
extern "C" void kernel_launch(void* const* d_in, const int* in_sizes, int n_in,
                              void* d_out, int out_size, void* d_ws, size_t ws_size,
                              hipStream_t stream) {
  const float* x  = (const float*)d_in[0];   // [B,T,D]
  const float* wa = (const float*)d_in[1];   // [D,3D]
  const float* wp = (const float*)d_in[2];   // [D,D]
  float* out = (float*)d_out;                // [B,T,D] fp32

  const size_t M    = (size_t)BATCH * SEQ;            // 8192
  const size_t nX   = M * D_MODEL;                    // 8,388,608
  const size_t nWA  = (size_t)D_MODEL * 3 * D_MODEL;  // 3,145,728
  const size_t nWP  = (size_t)D_MODEL * D_MODEL;      // 1,048,576
  const size_t nQKV = M * 3 * D_MODEL;                // 25,165,824
  const size_t nVT  = nX;                             // VT [B][H][hd][T]

  __bf16* xb    = (__bf16*)d_ws;           // x bf16          [M][K]
  __bf16* wabT  = xb    + nX;              // W_attn^T bf16   [3D][D]
  __bf16* wpbT  = wabT  + nWA;             // W_proj^T bf16   [D][D]
  __bf16* qkvb  = wpbT  + nWP;             // qkv bf16        [M][3D]
  __bf16* vtb   = qkvb  + nQKV;            // V^T bf16        [B][H][hd][T]
  __bf16* yb    = vtb   + nVT;             // attn out bf16   [M][D]

  cvt_bf16_kernel<<<dim3((nX / 4 + 255) / 256), 256, 0, stream>>>(x, xb, (int)(nX / 4));
  cvt_transpose_bf16_kernel<<<dim3((nWA / 8 + 255) / 256), 256, 0, stream>>>(
      wa, wabT, D_MODEL, 3 * D_MODEL);
  cvt_transpose_bf16_kernel<<<dim3((nWP / 8 + 255) / 256), 256, 0, stream>>>(
      wp, wpbT, D_MODEL, D_MODEL);

  // qkv = x @ W_attn : [8192,3072]
  dim3 g1(3 * D_MODEL / 128, (unsigned)(M / 128));
  gemm_bf16_wmma<false><<<g1, 128, 0, stream>>>(xb, wabT, (void*)qkvb,
                                                (int)M, 3 * D_MODEL, D_MODEL);

  // repack V -> VT (keys contiguous)
  repack_vt_kernel<<<dim3((nVT / 8) / 256), 256, 0, stream>>>(qkvb, vtb);

  // attention: 32 queries per wave, 128 per block
  flash_attn_wmma<<<dim3(BATCH * NHEAD * SEQ / 128), 128, 0, stream>>>(qkvb, vtb, yb);

  // out = y @ W_proj : [8192,1024] fp32
  dim3 g2(D_MODEL / 128, (unsigned)(M / 128));
  gemm_bf16_wmma<true><<<g2, 128, 0, stream>>>(yb, wpbT, (void*)out,
                                               (int)M, D_MODEL, D_MODEL);
}
